// MambaNet_22797686407366
// MI455X (gfx1250) — matrix operand order repaired and checked by reading, not compiled, not run on hardware
//
#include <hip/hip_runtime.h>
#include <hip/hip_bf16.h>
#include <stdint.h>

// ---------------------------------------------------------------------------
// Mamba (2-layer) forward for MI455X / gfx1250.
// GEMMs: v_wmma_f32_16x16x32_bf16 (wave32 WMMA, f32 accumulate).
// Selective scan: 3-phase chunk-parallel associative scan for occupancy,
// with async global->LDS staging of B_t/C_t in phase 3.
// ---------------------------------------------------------------------------

typedef __attribute__((ext_vector_type(16))) __bf16   v16bf;
typedef __attribute__((ext_vector_type(8)))  float    v8f;
typedef __attribute__((ext_vector_type(8)))  uint32_t v8u;

#define N_LAYERS 2
#define D_MODEL  1024
#define D_STATE  64
#define D_CONV   4
#define D_INNER  2048
#define DT_RANK  64
#define BATCH    2
#define SEQLEN   1024
#define MROWS    (BATCH * SEQLEN)          // 2048
#define XDBL_N   (DT_RANK + 2 * D_STATE)   // 192
#define CHUNK    128
#define NCHUNK   (SEQLEN / CHUNK)          // 8

#if defined(__gfx1250__) && __has_builtin(__builtin_amdgcn_global_load_async_to_lds_b32)
#define HAVE_ASYNC_LDS 1
#else
#define HAVE_ASYNC_LDS 0
#endif

__device__ __forceinline__ uint16_t f2bf(float f) {
  uint32_t u = __float_as_uint(f);
  u += 0x7FFFu + ((u >> 16) & 1u);         // round-to-nearest-even
  return (uint16_t)(u >> 16);
}
__device__ __forceinline__ float sigmoidf_(float x) {
  return 1.0f / (1.0f + __expf(-x));
}

// ---------------------------------------------------------------------------
// fp32 -> bf16 convert
// ---------------------------------------------------------------------------
__global__ void k_f32_to_bf16(const float* __restrict__ src,
                              uint16_t* __restrict__ dst, int n) {
  int i = blockIdx.x * blockDim.x + threadIdx.x;
  if (i < n) dst[i] = f2bf(src[i]);
}

// ---------------------------------------------------------------------------
// WMMA GEMM:  C[M,N] = A[M,K] * W[N,K]^T          (bf16 in, f32 out)
// Block = 128 threads = 4 waves; wave w covers rows [blockIdx.x*64 + w*16, +16)
// and cols [blockIdx.y*64, +64) via 4 B-fragments sharing one A-fragment.
// mode 0: plain store;  mode 1: softplus(acc + bias[n])  (dt_proj epilogue)
// ---------------------------------------------------------------------------
__global__ __launch_bounds__(128) void k_gemm_bf16(
    const uint16_t* __restrict__ A,   // [M,K] bf16 row-major
    const uint16_t* __restrict__ W,   // [N,K] bf16 row-major
    float* __restrict__ C,            // [M,N] f32
    int M, int N, int K,
    int mode, const float* __restrict__ bias) {
  const int lane = threadIdx.x & 31;
  const int wv   = threadIdx.x >> 5;
  const int l16  = lane & 15;
  const int hf   = lane >> 4;               // 0 = lanes 0-15, 1 = lanes 16-31
  const int mrow = (blockIdx.x * 4 + wv) * 16;
  const int n0   = blockIdx.y * 64;

  const uint32_t* Ap  = (const uint32_t*)(A + (size_t)(mrow + l16) * K);
  const uint32_t* Bp0 = (const uint32_t*)(W + (size_t)(n0      + l16) * K);
  const uint32_t* Bp1 = (const uint32_t*)(W + (size_t)(n0 + 16 + l16) * K);
  const uint32_t* Bp2 = (const uint32_t*)(W + (size_t)(n0 + 32 + l16) * K);
  const uint32_t* Bp3 = (const uint32_t*)(W + (size_t)(n0 + 48 + l16) * K);

  v8f acc0 = {}, acc1 = {}, acc2 = {}, acc3 = {};
  const int hb4 = hf * 4;                   // unused; A frag also uses hb8
  const int hb8 = hf * 8;                   // B-frag half offset (dwords)

  for (int kb = 0; kb < K; kb += 32) {
    const int base = kb >> 1;               // dword index into a bf16 row
    v8u ar, b0, b1, b2, b3;
#pragma unroll
    for (int v = 0; v < 8; ++v) {
      // A frag (this kernel's contiguous k order, same as B): VGPR v holds K = hf*16 + v*2 (+1)
      const int ia = base + (hb8 + v);
      // B frag (this kernel's contiguous k order, same as A): VGPR v holds K = hf*16 + v*2 (+1)
      const int ib = base + (hb8 + v);
      ar[v] = Ap[ia];
      b0[v] = Bp0[ib];
      b1[v] = Bp1[ib];
      b2[v] = Bp2[ib];
      b3[v] = Bp3[ib];
    }
    if (kb + 32 < K) {                       // global_prefetch_b8 of next tiles
      __builtin_prefetch(Ap  + (base + 16), 0, 0);
      __builtin_prefetch(Bp0 + (base + 16), 0, 0);
      __builtin_prefetch(Bp2 + (base + 16), 0, 0);
    }
    const v16bf av = __builtin_bit_cast(v16bf, ar);
    acc0 = __builtin_amdgcn_wmma_f32_16x16x32_bf16(
        false, av, false, __builtin_bit_cast(v16bf, b0), (short)0, acc0, false, false);
    acc1 = __builtin_amdgcn_wmma_f32_16x16x32_bf16(
        false, av, false, __builtin_bit_cast(v16bf, b1), (short)0, acc1, false, false);
    acc2 = __builtin_amdgcn_wmma_f32_16x16x32_bf16(
        false, av, false, __builtin_bit_cast(v16bf, b2), (short)0, acc2, false, false);
    acc3 = __builtin_amdgcn_wmma_f32_16x16x32_bf16(
        false, av, false, __builtin_bit_cast(v16bf, b3), (short)0, acc3, false, false);
  }

  // C 16x16 f32 layout: VGPR r -> m = mrow + r + hf*8, n = n0 + j*16 + (lane&15)
#pragma unroll
  for (int r = 0; r < 8; ++r) {
    const int m = mrow + hf * 8 + r;
    float vals[4] = {acc0[r], acc1[r], acc2[r], acc3[r]};
#pragma unroll
    for (int j = 0; j < 4; ++j) {
      const int n = n0 + j * 16 + l16;
      float v = vals[j];
      if (mode == 1) {                       // softplus(x + dt_proj_b[n])
        v += bias[n];
        v = (v > 20.0f) ? v : log1pf(__expf(v));
      }
      C[(size_t)m * N + n] = v;
    }
  }
}

// ---------------------------------------------------------------------------
// causal depthwise conv (K=4) + SiLU on u = xz[:, 0:2048]
// ---------------------------------------------------------------------------
__global__ void k_conv_silu(const float* __restrict__ xz,
                            const float* __restrict__ cw,
                            const float* __restrict__ cb,
                            float* __restrict__ uf,
                            uint16_t* __restrict__ ub) {
  int i = blockIdx.x * blockDim.x + threadIdx.x;
  if (i >= MROWS * D_INNER) return;
  const int c = i & (D_INNER - 1);
  const int m = i >> 11;
  const int l = m & (SEQLEN - 1);
  const int b = m >> 10;
  float acc = cb[c];
#pragma unroll
  for (int k = 0; k < D_CONV; ++k) {
    const int lk = l + k - (D_CONV - 1);
    if (lk >= 0)
      acc += xz[((size_t)(b * SEQLEN + lk)) * (2 * D_INNER) + c] * cw[c * D_CONV + k];
  }
  const float s = acc * sigmoidf_(acc);
  uf[i] = s;
  ub[i] = f2bf(s);
}

// ---------------------------------------------------------------------------
// extract dt-rank slice of x_dbl as bf16 (A-operand of dt_proj GEMM)
// ---------------------------------------------------------------------------
__global__ void k_dtr(const float* __restrict__ xdbl, uint16_t* __restrict__ dtr) {
  int i = blockIdx.x * blockDim.x + threadIdx.x;
  if (i >= MROWS * DT_RANK) return;
  const int m = i >> 6;
  const int j = i & (DT_RANK - 1);
  dtr[i] = f2bf(xdbl[(size_t)m * XDBL_N + j]);
}

// ---------------------------------------------------------------------------
// scan phase 1: per (b, chunk, c, s) compute chunk summary
//   P = prod_t a_t,   H = local h at chunk end (h=0 at chunk start)
//   a_t = exp(dt*A[c,s]),  w_t = dt*u*B_t[s]
// grid: (D_INNER/4, NCHUNK, BATCH), block 256 = 4 channels x 64 states
// ---------------------------------------------------------------------------
__global__ __launch_bounds__(256) void k_scan_phase1(
    const float* __restrict__ uf, const float* __restrict__ dt,
    const float* __restrict__ xdbl, const float* __restrict__ A_log,
    float* __restrict__ chunkP, float* __restrict__ chunkH) {
  const int tid = threadIdx.x;
  const int s   = tid & (D_STATE - 1);
  const int c   = blockIdx.x * 4 + (tid >> 6);
  const int j   = blockIdx.y;
  const int b   = blockIdx.z;
  const float Acs = -__expf(A_log[(size_t)c * D_STATE + s]);
  float P = 1.0f, H = 0.0f;
  for (int t = j * CHUNK; t < (j + 1) * CHUNK; ++t) {
    const size_t m  = (size_t)b * SEQLEN + t;
    const float dtv = dt[m * D_INNER + c];
    const float uv  = uf[m * D_INNER + c];
    const float Bv  = xdbl[m * XDBL_N + DT_RANK + s];
    const float a   = __expf(dtv * Acs);
    P *= a;
    H = a * H + dtv * uv * Bv;
  }
  const size_t o = (((size_t)b * NCHUNK + j) * D_INNER + c) * D_STATE + s;
  chunkP[o] = P;
  chunkH[o] = H;
}

// ---------------------------------------------------------------------------
// scan phase 2: chunk-level recurrence, h_start per chunk
// ---------------------------------------------------------------------------
__global__ void k_scan_phase2(const float* __restrict__ chunkP,
                              const float* __restrict__ chunkH,
                              float* __restrict__ hstart) {
  int i = blockIdx.x * blockDim.x + threadIdx.x;   // over BATCH*D_INNER*D_STATE
  if (i >= BATCH * D_INNER * D_STATE) return;
  const int b  = i / (D_INNER * D_STATE);
  const int cs = i % (D_INNER * D_STATE);
  float h = 0.0f;
#pragma unroll
  for (int j = 0; j < NCHUNK; ++j) {
    const size_t o = ((size_t)(b * NCHUNK + j)) * D_INNER * D_STATE + cs;
    hstart[o] = h;
    h = chunkP[o] * h + chunkH[o];
  }
}

// ---------------------------------------------------------------------------
// scan phase 3: replay within chunk from h_start, produce y
// grid: (D_INNER/256, NCHUNK, BATCH), block 256 (one channel per thread)
// B_t|C_t (xdbl cols 64..191, contiguous) staged via async global->LDS.
// ---------------------------------------------------------------------------
__global__ __launch_bounds__(256) void k_scan_phase3(
    const float* __restrict__ uf, const float* __restrict__ dt,
    const float* __restrict__ xdbl, const float* __restrict__ A_log,
    const float* __restrict__ Dsk, const float* __restrict__ hstart,
    float* __restrict__ y) {
  __shared__ float sBC[2 * D_STATE];
  const int tid = threadIdx.x;
  const int c   = blockIdx.x * 256 + tid;
  const int j   = blockIdx.y;
  const int b   = blockIdx.z;

  float Arow[D_STATE], h[D_STATE];
  const size_t hoff = (((size_t)b * NCHUNK + j) * D_INNER + c) * D_STATE;
#pragma unroll
  for (int s = 0; s < D_STATE; ++s) {
    Arow[s] = -__expf(A_log[(size_t)c * D_STATE + s]);
    h[s] = hstart[hoff + s];
  }
  const float Dc = Dsk[c];

  for (int t = j * CHUNK; t < (j + 1) * CHUNK; ++t) {
    const size_t m = (size_t)b * SEQLEN + t;
#if HAVE_ASYNC_LDS
    if (tid < 2 * D_STATE) {
      const float* g = xdbl + m * XDBL_N + DT_RANK + tid;
      __builtin_amdgcn_global_load_async_to_lds_b32(
          (__attribute__((address_space(1))) int*)(uintptr_t)g,
          (__attribute__((address_space(3))) int*)(sBC + tid),
          0, 0);
    }
    asm volatile("s_wait_asynccnt 0x0" ::: "memory");
    __syncthreads();
#else
    if (tid < 2 * D_STATE)
      sBC[tid] = xdbl[m * XDBL_N + DT_RANK + tid];
    __syncthreads();
#endif
    const float dtv = dt[m * D_INNER + c];
    const float uv  = uf[m * D_INNER + c];
    const float du  = dtv * uv;
    float acc = uv * Dc;
#pragma unroll
    for (int s = 0; s < D_STATE; ++s) {
      h[s] = h[s] * __expf(dtv * Arow[s]) + du * sBC[s];
      acc += h[s] * sBC[D_STATE + s];
    }
    y[m * D_INNER + c] = acc;
    __syncthreads();
  }
}

// ---------------------------------------------------------------------------
// y *= silu(z);  also emit bf16 copy for out_proj GEMM
// ---------------------------------------------------------------------------
__global__ void k_gate(const float* __restrict__ xz,
                       float* __restrict__ y, uint16_t* __restrict__ yb) {
  int i = blockIdx.x * blockDim.x + threadIdx.x;
  if (i >= MROWS * D_INNER) return;
  const int c = i & (D_INNER - 1);
  const int m = i >> 11;
  const float z = xz[(size_t)m * (2 * D_INNER) + D_INNER + c];
  const float v = y[i] * (z * sigmoidf_(z));
  y[i] = v;
  yb[i] = f2bf(v);
}

// ---------------------------------------------------------------------------
// host-side orchestration
// ---------------------------------------------------------------------------
extern "C" void kernel_launch(void* const* d_in, const int* in_sizes, int n_in,
                              void* d_out, int out_size, void* d_ws, size_t ws_size,
                              hipStream_t stream) {
  const float* x_in  = (const float*)d_in[0];
  const float* w_in  = (const float*)d_in[1];   // [L, 4096, 1024]
  const float* cw    = (const float*)d_in[2];   // [L, 2048, 4]
  const float* cb    = (const float*)d_in[3];   // [L, 2048]
  const float* w_xp  = (const float*)d_in[4];   // [L, 192, 2048]
  const float* w_dt  = (const float*)d_in[5];   // [L, 2048, 64]
  const float* b_dt  = (const float*)d_in[6];   // [L, 2048]
  const float* A_log = (const float*)d_in[7];   // [L, 2048, 64]
  const float* Dsk   = (const float*)d_in[8];   // [L, 2048]
  const float* w_out = (const float*)d_in[9];   // [L, 1024, 2048]
  float* out = (float*)d_out;

  char*  ws  = (char*)d_ws;
  size_t off = 0;
  auto carve = [&](size_t bytes) -> char* {
    char* p = ws + off;
    off = (off + bytes + 255) & ~(size_t)255;
    return p;
  };
  uint16_t* xbf    = (uint16_t*)carve((size_t)MROWS * D_MODEL * 2);
  uint16_t* winb   = (uint16_t*)carve((size_t)2 * D_INNER * D_MODEL * 2);
  uint16_t* wxpb   = (uint16_t*)carve((size_t)XDBL_N * D_INNER * 2);
  uint16_t* wdtb   = (uint16_t*)carve((size_t)D_INNER * DT_RANK * 2);
  uint16_t* woutb  = (uint16_t*)carve((size_t)D_MODEL * D_INNER * 2);
  float*    xz     = (float*)carve((size_t)MROWS * 2 * D_INNER * 4);
  float*    uf     = (float*)carve((size_t)MROWS * D_INNER * 4);
  uint16_t* ub     = (uint16_t*)carve((size_t)MROWS * D_INNER * 2);
  float*    xdbl   = (float*)carve((size_t)MROWS * XDBL_N * 4);
  uint16_t* dtrb   = (uint16_t*)carve((size_t)MROWS * DT_RANK * 2);
  float*    dtf    = (float*)carve((size_t)MROWS * D_INNER * 4);
  float*    yf     = (float*)carve((size_t)MROWS * D_INNER * 4);
  uint16_t* ybf    = (uint16_t*)carve((size_t)MROWS * D_INNER * 2);
  float*    xnext  = (float*)carve((size_t)MROWS * D_MODEL * 4);
  float*    chunkP = (float*)carve((size_t)BATCH * NCHUNK * D_INNER * D_STATE * 4);
  float*    chunkH = (float*)carve((size_t)BATCH * NCHUNK * D_INNER * D_STATE * 4);
  float*    hstart = (float*)carve((size_t)BATCH * NCHUNK * D_INNER * D_STATE * 4);
  (void)in_sizes; (void)n_in; (void)out_size; (void)ws_size;

  auto cvt = [&](const float* s, uint16_t* d, int n) {
    k_f32_to_bf16<<<(n + 255) / 256, 256, 0, stream>>>(s, d, n);
  };

  cvt(x_in, xbf, MROWS * D_MODEL);

  for (int layer = 0; layer < N_LAYERS; ++layer) {
    const float* Al = A_log + (size_t)layer * D_INNER * D_STATE;

    cvt(w_in  + (size_t)layer * 2 * D_INNER * D_MODEL, winb,  2 * D_INNER * D_MODEL);
    cvt(w_xp  + (size_t)layer * XDBL_N * D_INNER,      wxpb,  XDBL_N * D_INNER);
    cvt(w_dt  + (size_t)layer * D_INNER * DT_RANK,     wdtb,  D_INNER * DT_RANK);
    cvt(w_out + (size_t)layer * D_MODEL * D_INNER,     woutb, D_MODEL * D_INNER);

    // in_proj: [2048,1024] x [4096,1024]^T -> xz [2048,4096]
    k_gemm_bf16<<<dim3(MROWS / 64, (2 * D_INNER) / 64), 128, 0, stream>>>(
        xbf, winb, xz, MROWS, 2 * D_INNER, D_MODEL, 0, nullptr);

    // causal conv + SiLU -> u (f32 + bf16)
    k_conv_silu<<<(MROWS * D_INNER + 255) / 256, 256, 0, stream>>>(
        xz, cw + (size_t)layer * D_INNER * D_CONV, cb + (size_t)layer * D_INNER, uf, ub);

    // x_proj: [2048,2048] x [192,2048]^T -> x_dbl [2048,192]
    k_gemm_bf16<<<dim3(MROWS / 64, XDBL_N / 64), 128, 0, stream>>>(
        ub, wxpb, xdbl, MROWS, XDBL_N, D_INNER, 0, nullptr);

    // dt-rank slice to bf16
    k_dtr<<<(MROWS * DT_RANK + 255) / 256, 256, 0, stream>>>(xdbl, dtrb);

    // dt_proj + fused softplus(x + b): [2048,64] x [2048,64]^T -> dt [2048,2048]
    k_gemm_bf16<<<dim3(MROWS / 64, D_INNER / 64), 128, 0, stream>>>(
        dtrb, wdtb, dtf, MROWS, D_INNER, DT_RANK, 1, b_dt + (size_t)layer * D_INNER);

    // chunk-parallel selective scan
    k_scan_phase1<<<dim3(D_INNER / 4, NCHUNK, BATCH), 256, 0, stream>>>(
        uf, dtf, xdbl, Al, chunkP, chunkH);
    k_scan_phase2<<<(BATCH * D_INNER * D_STATE + 255) / 256, 256, 0, stream>>>(
        chunkP, chunkH, hstart);
    k_scan_phase3<<<dim3(D_INNER / 256, NCHUNK, BATCH), 256, 0, stream>>>(
        uf, dtf, xdbl, Al, Dsk + (size_t)layer * D_INNER, hstart, yf);

    // gate: y *= silu(z)
    k_gate<<<(MROWS * D_INNER + 255) / 256, 256, 0, stream>>>(xz, yf, ybf);

    // out_proj: [2048,2048] x [1024,2048]^T -> [2048,1024]
    float* dst = (layer == N_LAYERS - 1) ? out : xnext;
    k_gemm_bf16<<<dim3(MROWS / 64, D_MODEL / 64), 128, 0, stream>>>(
        ybf, woutb, dst, MROWS, D_MODEL, D_INNER, 0, nullptr);

    if (layer != N_LAYERS - 1) cvt(xnext, xbf, MROWS * D_MODEL);
  }
}
